// TemporalFraudDetector_49383533969490
// MI455X (gfx1250) — compile-verified, matrix operand order checked by reading, not run on hardware
//
#include <hip/hip_runtime.h>
#include <hip/hip_bf16.h>
#include <math.h>

#define TT 4
#define NT 20000
#define NC 20000
#define NE 10000
#define FIN 64
#define HID 128

typedef __attribute__((ext_vector_type(16))) __bf16 v16bf;
typedef __attribute__((ext_vector_type(8)))  __bf16 v8bf;
typedef __attribute__((ext_vector_type(8)))  float  v8f;

__device__ __forceinline__ unsigned short bf16_bits(float x) {
    // round-to-nearest-even f32 -> bf16 bits
    unsigned u = __float_as_uint(x);
    unsigned r = (u + 0x7FFFu + ((u >> 16) & 1u)) >> 16;
    return (unsigned short)r;
}

__device__ __forceinline__ float sigmoidf_(float x) { return 1.0f / (1.0f + expf(-x)); }

// K index for the 16-bit 16x32 WMMA A/B layout (ISA 7.12.2): element e (0..15)
// of the per-lane v16bf, lane-half hi = (lane>=16).
__device__ __forceinline__ int wmma_k_of(int hi, int e) {
    int v = e >> 1, h = e & 1;
    return ((v >> 2) << 4) + (hi << 3) + ((v & 3) << 1) + h;
}

// ---------------------------------------------------------------------------
// Weight pre-pack: B panel (K x N, row-major or transposed) -> bf16 in exact
// WMMA lane order.  Layout: [nchunk(N/128)][kc(K/32)][nt(8)][lane(32)][e(16)]
// so one block's panel (all kc, 8 n-tiles) is contiguous for LDS staging.
// ---------------------------------------------------------------------------
__global__ void pack_weights_kernel(const float* __restrict__ B, int ldb, int btrans,
                                    unsigned short* __restrict__ P, int K, int N)
{
    int idx = blockIdx.x * blockDim.x + threadIdx.x;
    if (idx >= K * N) return;
    int e    = idx & 15;
    int lane = (idx >> 4) & 31;
    int tile = idx >> 9;                 // global tile index
    int kchunks = K >> 5;
    int nchunk  = tile / (kchunks * 8);
    int rem     = tile % (kchunks * 8);
    int kc      = (rem >> 3) << 5;
    int nt      = rem & 7;
    int n  = (nchunk << 7) + (nt << 4) + (lane & 15);
    int hi = (lane >> 4) & 1;
    int k  = kc + wmma_k_of(hi, e);
    float w = btrans ? B[(size_t)n * ldb + k] : B[(size_t)k * ldb + n];
    P[idx] = bf16_bits(w);
}

__global__ void f32_to_bf16_kernel(const float* __restrict__ src,
                                   unsigned short* __restrict__ dst, int n)
{
    int i = blockIdx.x * blockDim.x + threadIdx.x;
    if (i < n) dst[i] = bf16_bits(src[i]);
}

// ---------------------------------------------------------------------------
// WMMA GEMM: Y[M, N] = X[M, K] @ B (+bias, optional relu). N via gridDim.y*128.
// X is row-major bf16 (a lane's A fragment = two contiguous 8-elem K-runs, so
// row-major bf16 needs no in-loop conversion). B pre-packed bf16, staged via
// LDS. One wave computes a 16x128 tile (8 wmma n-tiles), K fully unrolled.
// ---------------------------------------------------------------------------
template <int K>
__global__ __launch_bounds__(256) void gemm_bf16_wmma(
    const unsigned short* __restrict__ X, int ldx,
    const unsigned short* __restrict__ P,
    const float* __restrict__ bias,
    float* __restrict__ Y, int ldy,
    int M, int relu_flag)
{
    constexpr int KC = K / 32;
    __shared__ __attribute__((aligned(32))) unsigned short Bs[KC * 8 * 512];

    // stage this block's packed panel (n-chunk = blockIdx.y) into LDS
    {
        const uint4* src = (const uint4*)(P + (size_t)blockIdx.y * (KC * 8 * 512));
        uint4* dst = (uint4*)Bs;
        const int nv = KC * 8 * 512 * 2 / 16;
#pragma unroll
        for (int i = 0; i < nv / 256; ++i) dst[i * 256 + threadIdx.x] = src[i * 256 + threadIdx.x];
    }
    __syncthreads();

    const int lane  = threadIdx.x & 31;
    const int wave  = threadIdx.x >> 5;
    const int mtile = blockIdx.x * 8 + wave;
    const int m0    = mtile * 16;
    if (m0 >= M) return;                 // wave-uniform exit (EXEC all-ones for WMMA)
    const int n0  = blockIdx.y * 128;
    const int row = lane & 15;
    const int hi  = (lane >> 4) & 1;

    v8f acc[8];
#pragma unroll
    for (int i = 0; i < 8; ++i)
#pragma unroll
        for (int j = 0; j < 8; ++j) acc[i][j] = 0.0f;

    // A row: clamp instead of zero-fill (an A row only feeds its own D row;
    // rows >= M are never stored).
    const int mrow = (m0 + row < M) ? (m0 + row) : (M - 1);
    const unsigned short* xr = X + (size_t)mrow * ldx;
    const v16bf* bs = (const v16bf*)Bs;

#pragma unroll
    for (int kcI = 0; kcI < KC; ++kcI) {
        const int kc = kcI * 32;
        // lane's 16 A elements are two contiguous 8-bf16 (16B) runs:
        //   e=0..7  -> k = kc + hi*8 + e
        //   e=8..15 -> k = kc + 16 + hi*8 + (e-8)
        const v8bf r0 = *(const v8bf*)(xr + kc + hi * 8);
        const v8bf r1 = *(const v8bf*)(xr + kc + 16 + hi * 8);
        const v16bf a = __builtin_shufflevector(r0, r1, 0, 1, 2, 3, 4, 5, 6, 7,
                                                8, 9, 10, 11, 12, 13, 14, 15);
#pragma unroll
        for (int nt = 0; nt < 8; ++nt) {
            v16bf b = bs[((kcI * 8 + nt) << 5) + lane];   // 32B LDS read
            acc[nt] = __builtin_amdgcn_wmma_f32_16x16x32_bf16(
                false, a, false, b, (short)0, acc[nt], false, false);
        }
    }

#pragma unroll
    for (int nt = 0; nt < 8; ++nt) {
        int n = n0 + nt * 16 + row;
        float bv = bias ? bias[n] : 0.0f;
#pragma unroll
        for (int v = 0; v < 8; ++v) {
            int mo = m0 + v + (hi << 3);
            if (mo < M) {
                float r = acc[nt][v] + bv;
                if (relu_flag) r = fmaxf(r, 0.0f);
                Y[(size_t)mo * ldy + n] = r;
            }
        }
    }
}

// ---------------------------------------------------------------------------
// Elementwise / graph kernels
// ---------------------------------------------------------------------------
__global__ void deg_kernel(const int* __restrict__ ci, const int* __restrict__ ei,
                           float* degc, float* dege, int M)
{
    int i = blockIdx.x * blockDim.x + threadIdx.x;
    if (i >= M) return;
    atomicAdd(degc + ci[i], 1.0f);
    atomicAdd(dege + ei[i], 1.0f);
}

__global__ void scatter_add_kernel(const float* __restrict__ src, const int* __restrict__ idx,
                                   float* acc, int M)
{
    int t = blockIdx.x * blockDim.x + threadIdx.x;
    if (t >= M * HID) return;
    int i = t >> 7, c = t & 127;
    atomicAdd(acc + (size_t)idx[i] * HID + c, src[t]);
}

// v = relu(root[i,c] + Cbuf[ci[i],c] + Ebuf[ei[i],c] + bias[c]);
// optional bf16 activation output, optional pool-max
__global__ void combine_trans_kernel(const float* __restrict__ root,
                                     const float* __restrict__ Cbuf,
                                     const float* __restrict__ Ebuf,
                                     const int* __restrict__ ci, const int* __restrict__ ei,
                                     const float* __restrict__ bias,
                                     unsigned short* out_bf, float* poolmax, int M)
{
    int t = blockIdx.x * blockDim.x + threadIdx.x;
    if (t >= M * HID) return;
    int i = t >> 7, c = t & 127;
    float v = root[t] + Cbuf[(size_t)ci[i] * HID + c] + Ebuf[(size_t)ei[i] * HID + c] + bias[c];
    v = fmaxf(v, 0.0f);
    if (out_bf) out_bf[t] = bf16_bits(v);
    if (poolmax) atomicMax((unsigned int*)(poolmax + c), __float_as_uint(v));
}

// v = relu(root[i,c] + (deg[i]>0 ? acc[i,c]/deg[i] : 0) + bias[c]);
// optional bf16 activation output, optional pool-max
__global__ void finalize_mean_kernel(const float* __restrict__ root,
                                     const float* __restrict__ acc,
                                     const float* __restrict__ deg,
                                     const float* __restrict__ bias,
                                     unsigned short* out_bf, float* poolmax, int M)
{
    int t = blockIdx.x * blockDim.x + threadIdx.x;
    if (t >= M * HID) return;
    int i = t >> 7, c = t & 127;
    float d = deg[i];
    float a = (d > 0.0f) ? acc[t] / d : 0.0f;
    float v = fmaxf(root[t] + a + bias[c], 0.0f);
    if (out_bf) out_bf[t] = bf16_bits(v);
    if (poolmax) atomicMax((unsigned int*)(poolmax + c), __float_as_uint(v));
}

// length-1 GRU step with h0=0: h = (1-z)*n  (bf16 activation out)
__global__ void gru1_gate_kernel(const float* __restrict__ gi,
                                 const float* __restrict__ bih, const float* __restrict__ bhh,
                                 unsigned short* __restrict__ h, int M)
{
    int t = blockIdx.x * blockDim.x + threadIdx.x;
    if (t >= M * HID) return;
    int i = t >> 7, c = t & 127;
    const float* g = gi + (size_t)i * 384;
    float r = sigmoidf_(g[c]       + bih[c]       + bhh[c]);
    float z = sigmoidf_(g[128 + c] + bih[128 + c] + bhh[128 + c]);
    float n = tanhf(g[256 + c] + bih[256 + c] + r * bhh[256 + c]);
    h[t] = bf16_bits((1.0f - z) * n);
}

__global__ void row_dot_kernel(const float* __restrict__ lo, const float* __restrict__ w,
                               float* __restrict__ dots, int M)
{
    int i = blockIdx.x * blockDim.x + threadIdx.x;
    if (i >= M) return;
    const float* p = lo + (size_t)i * HID;
    float s = 0.0f;
#pragma unroll 8
    for (int c = 0; c < HID; ++c) s += p[c] * w[c];
    dots[i] = s;
}

// Tiny sequential GLU GRU (4 steps, batch 1) + fc + dot with final_W[0:128]
__global__ __launch_bounds__(128) void glu_temporal_kernel(
    const float* __restrict__ pooled,
    const float* __restrict__ Wih, const float* __restrict__ Whh,
    const float* __restrict__ bih, const float* __restrict__ bhh,
    const float* __restrict__ fcW, const float* __restrict__ fcb,
    const float* __restrict__ finW, float* __restrict__ s_out)
{
    __shared__ float hs[HID];
    __shared__ float xs[HID];
    __shared__ float red[HID];
    int c = threadIdx.x;
    hs[c] = 0.0f;
    __syncthreads();
    for (int t = 0; t < TT; ++t) {
        xs[c] = pooled[t * HID + c];
        __syncthreads();
        float gr = bih[c], gz = bih[128 + c], gn = bih[256 + c];
        float hr = bhh[c], hz = bhh[128 + c], hn = bhh[256 + c];
        for (int k = 0; k < HID; ++k) {
            float xv = xs[k], hv = hs[k];
            gr += xv * Wih[c * HID + k];
            gz += xv * Wih[(128 + c) * HID + k];
            gn += xv * Wih[(256 + c) * HID + k];
            hr += hv * Whh[c * HID + k];
            hz += hv * Whh[(128 + c) * HID + k];
            hn += hv * Whh[(256 + c) * HID + k];
        }
        float r = sigmoidf_(gr + hr);
        float z = sigmoidf_(gz + hz);
        float n = tanhf(gn + r * hn);
        float hnew = (1.0f - z) * n + z * hs[c];
        __syncthreads();
        hs[c] = hnew;
        __syncthreads();
        float y = fcb[c];
        for (int k = 0; k < HID; ++k) y += hs[k] * fcW[k * HID + c];
        red[c] = y * finW[c];
        __syncthreads();
        for (int off = 64; off > 0; off >>= 1) {
            if (c < off) red[c] += red[c + off];
            __syncthreads();
        }
        if (c == 0) s_out[t] = red[0];
        __syncthreads();
    }
}

__global__ void final_pred_kernel(const float* __restrict__ dots, const float* __restrict__ s,
                                  const float* __restrict__ fb, float* __restrict__ preds,
                                  int total)
{
    int j = blockIdx.x * blockDim.x + threadIdx.x;
    if (j >= total) return;
    int t = j / NT;
    preds[j] = sigmoidf_(s[t] + dots[j] + fb[0]);
}

// ---------------------------------------------------------------------------
// Host orchestration
// ---------------------------------------------------------------------------
extern "C" void kernel_launch(void* const* d_in, const int* in_sizes, int n_in,
                              void* d_out, int out_size, void* d_ws, size_t ws_size,
                              hipStream_t stream)
{
    const float* trans_x   = (const float*)d_in[0];
    const int*   card_idx  = (const int*)  d_in[1];
    const int*   email_idx = (const int*)  d_in[2];
    const float* labels    = (const float*)d_in[3];
    const float* card_emb  = (const float*)d_in[4];
    const float* email_emb = (const float*)d_in[5];
    const float* g1Wrel    = (const float*)d_in[6];   // [4,64,128]
    const float* g1Wroot   = (const float*)d_in[7];
    const float* g1b       = (const float*)d_in[8];
    const float* g2Wrel    = (const float*)d_in[9];   // [4,128,128]
    const float* g2Wroot   = (const float*)d_in[10];
    const float* g2b       = (const float*)d_in[11];
    const float* gWih      = (const float*)d_in[12];
    const float* gWhh      = (const float*)d_in[13];
    const float* gbih      = (const float*)d_in[14];
    const float* gbhh      = (const float*)d_in[15];
    const float* gfcW      = (const float*)d_in[16];
    const float* gfcb      = (const float*)d_in[17];
    const float* lWrel     = (const float*)d_in[18];  // [4,64,128]
    const float* lWroot    = (const float*)d_in[19];
    const float* lb        = (const float*)d_in[20];
    const float* lWih      = (const float*)d_in[21];
    const float* lbih      = (const float*)d_in[23];
    const float* lbhh      = (const float*)d_in[24];
    const float* lfcW      = (const float*)d_in[25];
    const float* lfcb      = (const float*)d_in[26];
    const float* finW      = (const float*)d_in[27];  // [256,1]
    const float* finb      = (const float*)d_in[28];

    // ---- workspace layout: float region, then bf16 (ushort) region
    float* W = (float*)d_ws;
    size_t o = 0;
    auto alloc = [&](size_t n) { float* p = W + o; o += n; return p; };
    float* C2g  = alloc((size_t)NC * HID);
    float* Crg  = alloc((size_t)NC * HID);
    float* E3g  = alloc((size_t)NE * HID);
    float* Erg  = alloc((size_t)NE * HID);
    float* C2l  = alloc((size_t)NC * HID);
    float* E3l  = alloc((size_t)NE * HID);
    float* A0   = alloc((size_t)NT * HID);
    float* A1   = alloc((size_t)NT * HID);
    float* Rt   = alloc((size_t)NT * HID);
    float* Rl   = alloc((size_t)NT * HID);
    float* accC = alloc((size_t)NC * HID);
    float* accE = alloc((size_t)NE * HID);
    float* degC = alloc(NC);
    float* degE = alloc(NE);
    float* C22  = alloc((size_t)NC * HID);
    float* Cr2  = alloc((size_t)NC * HID);
    float* E32  = alloc((size_t)NE * HID);
    float* Er2  = alloc((size_t)NE * HID);
    float* gi   = alloc((size_t)NT * 384);
    float* lo   = alloc((size_t)NT * HID);
    float* dots = alloc((size_t)TT * NT);
    float* pooled = alloc(TT * HID);
    float* sbuf   = alloc(TT);
    o = (o + 7) & ~(size_t)7;                       // 32B-align the ushort region
    unsigned short* U = (unsigned short*)(W + o);
    size_t uo = 0;
    auto ualloc = [&](size_t n) { unsigned short* p = U + uo; uo += n; return p; };
    // packed weights
    unsigned short* pG1R0 = ualloc(64 * 128);
    unsigned short* pG1R1 = ualloc(64 * 128);
    unsigned short* pG1R2 = ualloc(64 * 128);
    unsigned short* pG1R3 = ualloc(64 * 128);
    unsigned short* pG1Rt = ualloc(64 * 128);
    unsigned short* pLR2  = ualloc(64 * 128);
    unsigned short* pLR3  = ualloc(64 * 128);
    unsigned short* pLRt  = ualloc(64 * 128);
    unsigned short* pG2R0 = ualloc(128 * 128);
    unsigned short* pG2R1 = ualloc(128 * 128);
    unsigned short* pG2R2 = ualloc(128 * 128);
    unsigned short* pG2R3 = ualloc(128 * 128);
    unsigned short* pG2Rt = ualloc(128 * 128);
    unsigned short* pLWih = ualloc(128 * 384);
    unsigned short* pLfc  = ualloc(128 * 128);
    // bf16 activations (row-major, A-operand-ready)
    unsigned short* xbf  = ualloc((size_t)TT * NT * FIN);
    unsigned short* cbf  = ualloc((size_t)NC * FIN);
    unsigned short* ebf  = ualloc((size_t)NE * FIN);
    unsigned short* h1t  = ualloc((size_t)NT * HID);
    unsigned short* h1c  = ualloc((size_t)NC * HID);
    unsigned short* h1e  = ualloc((size_t)NE * HID);
    unsigned short* lt   = ualloc((size_t)NT * HID);
    unsigned short* hb   = ualloc((size_t)NT * HID);
    (void)ws_size; (void)in_sizes; (void)n_in; (void)out_size;

    const int EW = 256;
    auto blocksE = [&](int n) { return dim3((n + EW - 1) / EW); };
    auto pack = [&](const float* B, int ldb, int btrans, unsigned short* P, int K, int N) {
        pack_weights_kernel<<<blocksE(K * N), EW, 0, stream>>>(B, ldb, btrans, P, K, N);
    };
    auto cvt = [&](const float* src, unsigned short* dst, size_t n) {
        f32_to_bf16_kernel<<<blocksE((int)n), EW, 0, stream>>>(src, dst, (int)n);
    };
    auto gemm = [&](const unsigned short* X, int ldx, const unsigned short* P, const float* bias,
                    float* Y, int ldy, int M, int K, int N, int relu) {
        dim3 grid((M + 127) / 128, N / 128);
        if (K == 64)
            gemm_bf16_wmma<64><<<grid, dim3(256), 0, stream>>>(X, ldx, P, bias, Y, ldy, M, relu);
        else
            gemm_bf16_wmma<128><<<grid, dim3(256), 0, stream>>>(X, ldx, P, bias, Y, ldy, M, relu);
    };

    // ---- pack all weights + convert static inputs (cheap, every call) ----
    pack(g1Wrel + 0 * FIN * HID, HID, 0, pG1R0, 64, 128);
    pack(g1Wrel + 1 * FIN * HID, HID, 0, pG1R1, 64, 128);
    pack(g1Wrel + 2 * FIN * HID, HID, 0, pG1R2, 64, 128);
    pack(g1Wrel + 3 * FIN * HID, HID, 0, pG1R3, 64, 128);
    pack(g1Wroot,                HID, 0, pG1Rt, 64, 128);
    pack(lWrel + 2 * FIN * HID,  HID, 0, pLR2,  64, 128);
    pack(lWrel + 3 * FIN * HID,  HID, 0, pLR3,  64, 128);
    pack(lWroot,                 HID, 0, pLRt,  64, 128);
    pack(g2Wrel + 0 * HID * HID, HID, 0, pG2R0, 128, 128);
    pack(g2Wrel + 1 * HID * HID, HID, 0, pG2R1, 128, 128);
    pack(g2Wrel + 2 * HID * HID, HID, 0, pG2R2, 128, 128);
    pack(g2Wrel + 3 * HID * HID, HID, 0, pG2R3, 128, 128);
    pack(g2Wroot,                HID, 0, pG2Rt, 128, 128);
    pack(lWih,                   HID, 1, pLWih, 128, 384);   // gi = l @ Wih^T
    pack(lfcW,                   HID, 0, pLfc,  128, 128);
    cvt(trans_x,   xbf, (size_t)TT * NT * FIN);
    cvt(card_emb,  cbf, (size_t)NC * FIN);
    cvt(email_emb, ebf, (size_t)NE * FIN);

    // ---- time-invariant precompute ----
    gemm(cbf, FIN, pG1R2, nullptr, C2g, HID, NC, 64, 128, 0);
    gemm(cbf, FIN, pG1Rt, nullptr, Crg, HID, NC, 64, 128, 0);
    gemm(ebf, FIN, pG1R3, nullptr, E3g, HID, NE, 64, 128, 0);
    gemm(ebf, FIN, pG1Rt, nullptr, Erg, HID, NE, 64, 128, 0);
    gemm(cbf, FIN, pLR2,  nullptr, C2l, HID, NC, 64, 128, 0);
    gemm(ebf, FIN, pLR3,  nullptr, E3l, HID, NE, 64, 128, 0);
    hipMemsetAsync(pooled, 0, TT * HID * sizeof(float), stream);

    for (int t = 0; t < TT; ++t) {
        const unsigned short* x = xbf + (size_t)t * NT * FIN;
        const int* ci = card_idx  + (size_t)t * NT;
        const int* ei = email_idx + (size_t)t * NT;

        // ---- GLU layer 1 ----
        gemm(x, FIN, pG1R0, nullptr, A0, HID, NT, 64, 128, 0);
        gemm(x, FIN, pG1R1, nullptr, A1, HID, NT, 64, 128, 0);
        gemm(x, FIN, pG1Rt, nullptr, Rt, HID, NT, 64, 128, 0);
        gemm(x, FIN, pLRt,  nullptr, Rl, HID, NT, 64, 128, 0);

        hipMemsetAsync(accC, 0, (size_t)NC * HID * sizeof(float), stream);
        hipMemsetAsync(accE, 0, (size_t)NE * HID * sizeof(float), stream);
        hipMemsetAsync(degC, 0, NC * sizeof(float), stream);
        hipMemsetAsync(degE, 0, NE * sizeof(float), stream);
        deg_kernel<<<blocksE(NT), EW, 0, stream>>>(ci, ei, degC, degE, NT);
        scatter_add_kernel<<<blocksE(NT * HID), EW, 0, stream>>>(A0, ci, accC, NT);
        scatter_add_kernel<<<blocksE(NT * HID), EW, 0, stream>>>(A1, ei, accE, NT);

        combine_trans_kernel<<<blocksE(NT * HID), EW, 0, stream>>>(Rt, C2g, E3g, ci, ei, g1b, h1t, nullptr, NT);
        finalize_mean_kernel<<<blocksE(NC * HID), EW, 0, stream>>>(Crg, accC, degC, g1b, h1c, nullptr, NC);
        finalize_mean_kernel<<<blocksE(NE * HID), EW, 0, stream>>>(Erg, accE, degE, g1b, h1e, nullptr, NE);
        combine_trans_kernel<<<blocksE(NT * HID), EW, 0, stream>>>(Rl, C2l, E3l, ci, ei, lb, lt, nullptr, NT);

        // ---- GLU layer 2 (h2 fused into pool-max) ----
        gemm(h1t, HID, pG2R0, nullptr, A0,  HID, NT, 128, 128, 0);
        gemm(h1t, HID, pG2R1, nullptr, A1,  HID, NT, 128, 128, 0);
        gemm(h1t, HID, pG2Rt, nullptr, Rt,  HID, NT, 128, 128, 0);
        gemm(h1c, HID, pG2R2, nullptr, C22, HID, NC, 128, 128, 0);
        gemm(h1c, HID, pG2Rt, nullptr, Cr2, HID, NC, 128, 128, 0);
        gemm(h1e, HID, pG2R3, nullptr, E32, HID, NE, 128, 128, 0);
        gemm(h1e, HID, pG2Rt, nullptr, Er2, HID, NE, 128, 128, 0);

        hipMemsetAsync(accC, 0, (size_t)NC * HID * sizeof(float), stream);
        hipMemsetAsync(accE, 0, (size_t)NE * HID * sizeof(float), stream);
        scatter_add_kernel<<<blocksE(NT * HID), EW, 0, stream>>>(A0, ci, accC, NT);
        scatter_add_kernel<<<blocksE(NT * HID), EW, 0, stream>>>(A1, ei, accE, NT);

        float* pt = pooled + t * HID;
        combine_trans_kernel<<<blocksE(NT * HID), EW, 0, stream>>>(Rt, C22, E32, ci, ei, g2b, nullptr, pt, NT);
        finalize_mean_kernel<<<blocksE(NC * HID), EW, 0, stream>>>(Cr2, accC, degC, g2b, nullptr, pt, NC);
        finalize_mean_kernel<<<blocksE(NE * HID), EW, 0, stream>>>(Er2, accE, degE, g2b, nullptr, pt, NE);

        // ---- LLU temporal: 1-step GRU (h0=0) + fc + row-dot with final_W[128:256] ----
        gemm(lt, HID, pLWih, nullptr, gi, 384, NT, 128, 384, 0);
        gru1_gate_kernel<<<blocksE(NT * HID), EW, 0, stream>>>(gi, lbih, lbhh, hb, NT);
        gemm(hb, HID, pLfc, lfcb, lo, HID, NT, 128, 128, 0);
        row_dot_kernel<<<blocksE(NT), EW, 0, stream>>>(lo, finW + HID, dots + (size_t)t * NT, NT);
    }

    // ---- GLU temporal path (tiny) + final predictions ----
    glu_temporal_kernel<<<1, 128, 0, stream>>>(pooled, gWih, gWhh, gbih, gbhh, gfcW, gfcb, finW, sbuf);

    float* preds = (float*)d_out;
    final_pred_kernel<<<blocksE(TT * NT), EW, 0, stream>>>(dots, sbuf, finb, preds, TT * NT);

    hipMemcpyAsync(preds + (size_t)TT * NT, labels, (size_t)TT * NT * sizeof(float),
                   hipMemcpyDeviceToDevice, stream);
}
